// EGraphSAGE_44452911513780
// MI455X (gfx1250) — matrix operand order repaired and checked by reading, not compiled, not run on hardware
//
#include <hip/hip_runtime.h>
#include <hip/hip_bf16.h>

// ---------------------------------------------------------------------------
// EGraphSAGE edge-classification pipeline for MI455X (gfx1250, wave32, WMMA)
//
// All three GEMM stages on the matrix pipes (v_wmma_f32_16x16x32_f16):
//   0. zero agg/cnt
//   1. scatter-add edge_attr by dst            (global_atomic_add_f32)
//   2. agg = agg_sum / max(cnt,1)
//   3. h0 = LN(relu([x||agg] @ W0 + b0))       32 WMMAs / 16-node tile
//   4. z  = LN(relu([h0||agg] @ W1 + b1))      24 WMMAs / 16-node tile
//   5. out = relu([z[src]||z[dst]]@Wc0+bc0)@Wc1+bc1   16 WMMAs / 16-edge tile
//
// Fragment layouts used here (A and B share one contiguous K order):
//   A (16x32 f16): lane l -> row M=l&15; K-half=2*kbase, kbase=(l&16)?8:0,
//                  halves j: K = 2*kbase+j   [j=2v+o]
//   B (32x16 f16): lane l -> col N=l&15; K-half=(l&16)?16:0, K = half+j
//   D (16x16 f32): VGPR vv -> row M=vv+8*(l>=16); lane -> col N=l&15
// LayerNorm rides on the D layout: a row is one VGPR index across 16 lanes,
// so row stats are 4 shfl_xor butterflies.
// ---------------------------------------------------------------------------

typedef __attribute__((ext_vector_type(16))) _Float16 v16h;
typedef __attribute__((ext_vector_type(8)))  float    v8f;

#define NN   50000     // nodes
#define NE   800000    // edges
#define FD   64        // edge/node feature dim
#define H0D  128
#define H1D  64
#define EPSV 1e-5f

// ---------------- stage 0: zero scratch ------------------------------------
__global__ void k_zero(float* __restrict__ p, int n) {
    int i = blockIdx.x * blockDim.x + threadIdx.x;
    if (i < n) p[i] = 0.0f;
}

// ---------------- stage 1: scatter-add edge_attr by dst --------------------
__global__ void k_scatter(const float* __restrict__ ea,
                          const long long* __restrict__ ei,
                          float* __restrict__ aggsum,
                          float* __restrict__ cnt) {
    int idx = blockIdx.x * blockDim.x + threadIdx.x;   // over NE * FD
    int e = idx >> 6;
    int f = idx & 63;
    if (e >= NE) return;
    int d = (int)ei[NE + e];                           // dst row of edge_index
    atomicAdd(&aggsum[d * FD + f], ea[e * FD + f]);
    if (f == 0) atomicAdd(&cnt[d], 1.0f);
}

// ---------------- stage 2: divide by count ---------------------------------
__global__ void k_div(float* __restrict__ agg, const float* __restrict__ cnt) {
    int i = blockIdx.x * blockDim.x + threadIdx.x;
    if (i < NN * FD) {
        float c = cnt[i >> 6];
        agg[i] = agg[i] / fmaxf(c, 1.0f);
    }
}

// ---------------- helpers ---------------------------------------------------
// Build one A fragment (16x32 f16) from a row-major f32 row pointer p
// (this lane's row), K window = p[0..31].
__device__ __forceinline__ v16h load_afrag(const float* p, int kbase) {
    v16h a;
#pragma unroll
    for (int vv = 0; vv < 8; ++vv) {
        int K = 2 * kbase + 2 * vv;
        float2 f2 = *reinterpret_cast<const float2*>(p + K);
        a[2 * vv]     = (_Float16)f2.x;
        a[2 * vv + 1] = (_Float16)f2.y;
    }
    return a;
}

__device__ __forceinline__ float wave16_sum(float v) {
#pragma unroll
    for (int off = 1; off < 16; off <<= 1) v += __shfl_xor(v, off, 32);
    return v;
}

// ---------------- stage 3: node layer0 (WMMA, wave = 16 nodes) -------------
__global__ __launch_bounds__(256) void k_node0(
        const float* __restrict__ x,  const float* __restrict__ agg,
        const float* __restrict__ W0, const float* __restrict__ b0,
        const float* __restrict__ g0, const float* __restrict__ be0,
        float* __restrict__ h0) {
    // W0 [128x128] -> 8 N-tiles x 4 K-steps of B fragments (32 KB LDS)
    __shared__ __align__(32) _Float16 wfrag[32][32][16];
    int tid = threadIdx.x, lane = tid & 31, wid = tid >> 5;

    for (int i = tid; i < 32 * 32 * 16; i += 256) {
        int f = i >> 9, l = (i >> 4) & 31, j = i & 15;
        int nt = f >> 2, kc = f & 3;
        int ncol = nt * 16 + (l & 15);
        int K    = kc * 32 + ((l & 16) ? 16 : 0) + j;
        wfrag[f][l][j] = (_Float16)W0[K * H0D + ncol];
    }
    __syncthreads();

    int nl = lane & 15, kbase = (lane & 16) ? 8 : 0;
    float bv[8], gv[8], bev[8];
#pragma unroll
    for (int nt = 0; nt < 8; ++nt) {
        int ncol = nt * 16 + nl;
        bv[nt] = b0[ncol]; gv[nt] = g0[ncol]; bev[nt] = be0[ncol];
    }

    const int ntiles = NN / 16;          // 3125, exact
    int gw = blockIdx.x * 8 + wid, nw = gridDim.x * 8;

    for (int t = gw; t < ntiles; t += nw) {
        int node = t * 16 + nl;          // this lane's row (M = nl)
        const float* rx = x   + node * FD;
        const float* ra = agg + node * FD;

        v16h afr[4];
        afr[0] = load_afrag(rx,      kbase);
        afr[1] = load_afrag(rx + 32, kbase);
        afr[2] = load_afrag(ra,      kbase);
        afr[3] = load_afrag(ra + 32, kbase);

        v8f acc[8];
#pragma unroll
        for (int nt = 0; nt < 8; ++nt) {
            v8f c = {};
#pragma unroll
            for (int kc = 0; kc < 4; ++kc) {
                v16h b = *reinterpret_cast<const v16h*>(&wfrag[nt * 4 + kc][lane][0]);
                c = __builtin_amdgcn_wmma_f32_16x16x32_f16(
                        false, afr[kc], false, b, (short)0, c, false, false);
            }
            acc[nt] = c;
        }

        // bias + relu + layernorm per row, write normalized h0
#pragma unroll
        for (int vv = 0; vv < 8; ++vv) {
            float hv[8], s = 0.0f, q = 0.0f;
#pragma unroll
            for (int nt = 0; nt < 8; ++nt) {
                hv[nt] = fmaxf(acc[nt][vv] + bv[nt], 0.0f);
                s += hv[nt]; q = fmaf(hv[nt], hv[nt], q);
            }
            s = wave16_sum(s); q = wave16_sum(q);
            float m = s * (1.0f / H0D);
            float var = q * (1.0f / H0D) - m * m;
            float r = rsqrtf(var + EPSV);
            int row  = vv + ((lane & 16) ? 8 : 0);
            int nodeR = t * 16 + row;
#pragma unroll
            for (int nt = 0; nt < 8; ++nt)
                h0[nodeR * H0D + nt * 16 + nl] = (hv[nt] - m) * r * gv[nt] + bev[nt];
        }
    }
}

// ---------------- stage 4: node layer1 (WMMA, wave = 16 nodes) -------------
__global__ __launch_bounds__(256) void k_node1(
        const float* __restrict__ h0, const float* __restrict__ agg,
        const float* __restrict__ W1, const float* __restrict__ b1,
        const float* __restrict__ g1, const float* __restrict__ be1,
        float* __restrict__ z) {
    // W1 [192x64] -> 4 N-tiles x 6 K-steps of B fragments (24 KB LDS)
    __shared__ __align__(32) _Float16 wfrag[24][32][16];
    int tid = threadIdx.x, lane = tid & 31, wid = tid >> 5;

    for (int i = tid; i < 24 * 32 * 16; i += 256) {
        int f = i >> 9, l = (i >> 4) & 31, j = i & 15;
        int nt = f / 6, kc = f % 6;
        int ncol = nt * 16 + (l & 15);
        int K    = kc * 32 + ((l & 16) ? 16 : 0) + j;
        wfrag[f][l][j] = (_Float16)W1[K * H1D + ncol];
    }
    __syncthreads();

    int nl = lane & 15, kbase = (lane & 16) ? 8 : 0;
    float bv[4], gv[4], bev[4];
#pragma unroll
    for (int nt = 0; nt < 4; ++nt) {
        int ncol = nt * 16 + nl;
        bv[nt] = b1[ncol]; gv[nt] = g1[ncol]; bev[nt] = be1[ncol];
    }

    const int ntiles = NN / 16;
    int gw = blockIdx.x * 8 + wid, nw = gridDim.x * 8;

    for (int t = gw; t < ntiles; t += nw) {
        int node = t * 16 + nl;
        const float* rh = h0  + node * H0D;
        const float* ra = agg + node * FD;

        v16h afr[6];
#pragma unroll
        for (int kc = 0; kc < 4; ++kc) afr[kc] = load_afrag(rh + kc * 32, kbase);
        afr[4] = load_afrag(ra,      kbase);
        afr[5] = load_afrag(ra + 32, kbase);

        v8f acc[4];
#pragma unroll
        for (int nt = 0; nt < 4; ++nt) {
            v8f c = {};
#pragma unroll
            for (int kc = 0; kc < 6; ++kc) {
                v16h b = *reinterpret_cast<const v16h*>(&wfrag[nt * 6 + kc][lane][0]);
                c = __builtin_amdgcn_wmma_f32_16x16x32_f16(
                        false, afr[kc], false, b, (short)0, c, false, false);
            }
            acc[nt] = c;
        }

#pragma unroll
        for (int vv = 0; vv < 8; ++vv) {
            float hv[4], s = 0.0f, q = 0.0f;
#pragma unroll
            for (int nt = 0; nt < 4; ++nt) {
                hv[nt] = fmaxf(acc[nt][vv] + bv[nt], 0.0f);
                s += hv[nt]; q = fmaf(hv[nt], hv[nt], q);
            }
            s = wave16_sum(s); q = wave16_sum(q);
            float m = s * (1.0f / H1D);
            float var = q * (1.0f / H1D) - m * m;
            float r = rsqrtf(var + EPSV);
            int row  = vv + ((lane & 16) ? 8 : 0);
            int nodeR = t * 16 + row;
#pragma unroll
            for (int nt = 0; nt < 4; ++nt)
                z[nodeR * H1D + nt * 16 + nl] = (hv[nt] - m) * r * gv[nt] + bev[nt];
        }
    }
}

// ---------------- stage 5: edge classifier (WMMA, wave = 16 edges) ---------
__global__ __launch_bounds__(256) void k_edge_cls(
        const float* __restrict__ z, const long long* __restrict__ ei,
        const float* __restrict__ Wc0, const float* __restrict__ bc0,
        const float* __restrict__ Wc1, const float* __restrict__ bc1,
        float* __restrict__ out) {
    __shared__ __align__(32) _Float16 wfrag[16][32][16];
    int tid = threadIdx.x, lane = tid & 31, wid = tid >> 5;

    for (int i = tid; i < 16 * 32 * 16; i += 256) {
        int f = i >> 9, l = (i >> 4) & 31, j = i & 15;
        int nt = f >> 2, kc = f & 3;
        int ncol = nt * 16 + (l & 15);
        int K    = kc * 32 + ((l & 16) ? 16 : 0) + j;
        wfrag[f][l][j] = (_Float16)Wc0[K * H1D + ncol];
    }
    __syncthreads();

    int nl = lane & 15, kbase = (lane & 16) ? 8 : 0;
    float bc0v[4], wc1v[4][2];
#pragma unroll
    for (int nt = 0; nt < 4; ++nt) {
        int ncol = nt * 16 + nl;
        bc0v[nt]    = bc0[ncol];
        wc1v[nt][0] = Wc1[ncol * 2 + 0];
        wc1v[nt][1] = Wc1[ncol * 2 + 1];
    }
    float bo0 = bc1[0], bo1 = bc1[1];

    const int ntiles = NE / 16;            // 50000, exact
    int gw = blockIdx.x * 8 + wid, nw = gridDim.x * 8;

    for (int t = gw; t < ntiles; t += nw) {
        int e = t * 16 + nl;
        int sn = (int)ei[e];
        int dn = (int)ei[NE + e];
        const float* rs = z + sn * H1D;    // L2-resident (z = 12.8 MB)
        const float* rd = z + dn * H1D;

        v16h afr[4];
        afr[0] = load_afrag(rs,      kbase);
        afr[1] = load_afrag(rs + 32, kbase);
        afr[2] = load_afrag(rd,      kbase);
        afr[3] = load_afrag(rd + 32, kbase);

        v8f acc[4];
#pragma unroll
        for (int nt = 0; nt < 4; ++nt) {
            v8f c = {};
#pragma unroll
            for (int kc = 0; kc < 4; ++kc) {
                v16h b = *reinterpret_cast<const v16h*>(&wfrag[nt * 4 + kc][lane][0]);
                c = __builtin_amdgcn_wmma_f32_16x16x32_f16(
                        false, afr[kc], false, b, (short)0, c, false, false);
            }
            acc[nt] = c;
        }

        // relu + hidden[16x64] @ Wc1[64x2], reduced across each 16-lane half
#pragma unroll
        for (int vv = 0; vv < 8; ++vv) {
            float p0 = 0.0f, p1 = 0.0f;
#pragma unroll
            for (int nt = 0; nt < 4; ++nt) {
                float hvv = fmaxf(acc[nt][vv] + bc0v[nt], 0.0f);
                p0 = fmaf(hvv, wc1v[nt][0], p0);
                p1 = fmaf(hvv, wc1v[nt][1], p1);
            }
            p0 = wave16_sum(p0); p1 = wave16_sum(p1);
            if (nl == vv) {
                int m  = vv + ((lane & 16) ? 8 : 0);
                int eo = t * 16 + m;
                out[eo * 2 + 0] = p0 + bo0;
                out[eo * 2 + 1] = p1 + bo1;
            }
        }
    }
}

// ---------------------------------------------------------------------------
extern "C" void kernel_launch(void* const* d_in, const int* in_sizes, int n_in,
                              void* d_out, int out_size, void* d_ws, size_t ws_size,
                              hipStream_t stream) {
    (void)in_sizes; (void)n_in; (void)out_size; (void)ws_size;

    const float*     x   = (const float*)d_in[0];
    const long long* ei  = (const long long*)d_in[1];
    const float*     ea  = (const float*)d_in[2];
    const float*     W0  = (const float*)d_in[3];
    const float*     b0  = (const float*)d_in[4];
    const float*     g0  = (const float*)d_in[5];
    const float*     be0 = (const float*)d_in[6];
    const float*     W1  = (const float*)d_in[7];
    const float*     b1  = (const float*)d_in[8];
    const float*     g1  = (const float*)d_in[9];
    const float*     be1 = (const float*)d_in[10];
    const float*     Wc0 = (const float*)d_in[11];
    const float*     bc0 = (const float*)d_in[12];
    const float*     Wc1 = (const float*)d_in[13];
    const float*     bc1 = (const float*)d_in[14];
    float*           out = (float*)d_out;

    // workspace layout (256B-aligned chunks)
    char*  ws      = (char*)d_ws;
    size_t offAgg  = 0;
    size_t szAgg   = (size_t)NN * FD * sizeof(float);            // 12.8 MB
    size_t offCnt  = offAgg + szAgg;                             // contiguous
    size_t szCnt   = (size_t)NN * sizeof(float);
    size_t offH0   = (offCnt + szCnt + 255) & ~(size_t)255;
    size_t szH0    = (size_t)NN * H0D * sizeof(float);           // 25.6 MB
    size_t offZ    = (offH0 + szH0 + 255) & ~(size_t)255;        // 12.8 MB

    float* agg = (float*)(ws + offAgg);
    float* cnt = (float*)(ws + offCnt);
    float* h0  = (float*)(ws + offH0);
    float* zb  = (float*)(ws + offZ);

    // 0. zero agg_sum + cnt (contiguous NN*(FD+1) floats)
    {
        int n = NN * (FD + 1);
        k_zero<<<(n + 255) / 256, 256, 0, stream>>>(agg, n);
    }
    // 1. scatter-add
    {
        int n = NE * FD;
        k_scatter<<<(n + 255) / 256, 256, 0, stream>>>(ea, ei, agg, cnt);
    }
    // 2. divide
    {
        int n = NN * FD;
        k_div<<<(n + 255) / 256, 256, 0, stream>>>(agg, cnt);
    }
    // 3. node layer 0 (WMMA)
    k_node0<<<128, 256, 0, stream>>>(x, agg, W0, b0, g0, be0, h0);
    // 4. node layer 1 (WMMA)
    k_node1<<<128, 256, 0, stream>>>(h0, agg, W1, b1, g1, be1, zb);
    // 5. edge classifier (WMMA)
    k_edge_cls<<<1024, 256, 0, stream>>>(zb, ei, Wc0, bc0, Wc1, bc1, out);
}
